// PerformerAttention_8426725835196
// MI455X (gfx1250) — matrix operand order repaired and checked by reading, not compiled, not run on hardware
//
#include <hip/hip_runtime.h>

// ---------------------------------------------------------------------------
// Types for CDNA5 WMMA bf16: A/B fragments are 16 x bf16 per lane (wave32),
// C/D accumulator is 8 x f32 per lane.
// ---------------------------------------------------------------------------
typedef __attribute__((ext_vector_type(16))) __bf16 bf16x16;
typedef __attribute__((ext_vector_type(8)))  __bf16 bf16x8;
typedef __attribute__((ext_vector_type(8)))  float  f32x8;

union FragAB { bf16x16 v; bf16x8 h[2]; };

__device__ __forceinline__ unsigned short f32_to_bf16u(float f) {
  unsigned int u = __float_as_uint(f);
  u += 0x7FFFu + ((u >> 16) & 1u);          // round-to-nearest-even
  return (unsigned short)(u >> 16);
}

// CDNA5 async DMA: global -> LDS, 16B per lane, tracked by ASYNCcnt.
// VDST VGPR holds the wave-relative LDS byte address (== low 32 bits of a
// generic pointer to __shared__ memory, per the flat-aperture mapping).
__device__ __forceinline__ void async_b128(unsigned lds_off, const void* gptr) {
  asm volatile("global_load_async_to_lds_b128 %0, %1, off"
               :: "v"(lds_off), "v"((unsigned long long)(size_t)gptr)
               : "memory");
}
__device__ __forceinline__ void wait_async0() {
  asm volatile("s_wait_asynccnt 0" ::: "memory");
}
__device__ __forceinline__ unsigned lds_off_of(const void* p) {
  return (unsigned)(size_t)p;    // low 32 bits of generic LDS pointer
}

// ---------------------------------------------------------------------------
// bf16 WMMA GEMM:  C[M,N] = A[M,K] * Bt[N,K]^T  (B stored pre-transposed).
// Block tile 128x128, BK=64, double-buffered LDS filled by
// global_load_async_to_lds_b128. 8 waves in 4(M) x 2(N); each wave owns a
// 32x64 C tile = 8 accumulators; 8 WMMAs per 32-deep K-step.
// Epilogue: optional bias[N], f32 residual, relu, f32 and/or bf16 output.
// Batched via blockIdx.z + element strides. M,N multiples of 128; K of 64.
// ---------------------------------------------------------------------------
__global__ __launch_bounds__(256) void gemm_bf16_wmma(
    const __bf16* __restrict__ A, const __bf16* __restrict__ Bt,
    const float* __restrict__ bias, const float* __restrict__ resid,
    float* __restrict__ outF, unsigned short* __restrict__ outB,
    int M, int N, int K,
    long long strideA, long long strideB, long long strideC,
    int relu)
{
  // 72-half row stride (144B, 16B aligned) to spread LDS banks
  __shared__ __align__(16) __bf16 As[2][128][72];
  __shared__ __align__(16) __bf16 Bs[2][128][72];

  const int tid  = threadIdx.x;
  const int lane = tid & 31;
  const int wid  = tid >> 5;
  const int wm   = wid >> 1;        // 0..3 : 32-row slice
  const int wn   = wid & 1;         // 0..1 : 64-col slice
  const int hi   = lane >> 4;       // half-wave select
  const int lm   = lane & 15;

  const long long bm = (long long)blockIdx.y * 128;
  const long long bn = (long long)blockIdx.x * 128;
  const __bf16* Ab = A  + (long long)blockIdx.z * strideA;
  const __bf16* Bb = Bt + (long long)blockIdx.z * strideB;
  const long long cbase = (long long)blockIdx.z * strideC;

  f32x8 c[2][4];
  #pragma unroll
  for (int a = 0; a < 2; ++a)
    #pragma unroll
    for (int j = 0; j < 4; ++j)
      c[a][j] = f32x8{0.f,0.f,0.f,0.f,0.f,0.f,0.f,0.f};

  // DMA partition: 2 threads per row, 4x16B chunks each (row = 64 halves)
  const int grow = tid >> 1;                 // 0..127
  const int gcol = (tid & 1) * 32;           // 0 or 32 (halves)
  const __bf16* agp = Ab + (bm + grow) * (long long)K + gcol;
  const __bf16* bgp = Bb + (bn + grow) * (long long)K + gcol;

  const int nk = K >> 6;   // 64-deep stages

  // prologue: stage 0
  #pragma unroll
  for (int cch = 0; cch < 4; ++cch) {
    async_b128(lds_off_of(&As[0][grow][gcol + cch * 8]), agp + cch * 8);
    async_b128(lds_off_of(&Bs[0][grow][gcol + cch * 8]), bgp + cch * 8);
  }

  for (int i = 0; i < nk; ++i) {
    wait_async0();           // my stage-i DMAs done
    __syncthreads();         // everyone's DMAs done; prev compute finished

    if (i + 1 < nk) {
      const int nb = (i + 1) & 1;
      const long long ko = (long long)(i + 1) << 6;
      #pragma unroll
      for (int cch = 0; cch < 4; ++cch) {
        async_b128(lds_off_of(&As[nb][grow][gcol + cch * 8]), agp + ko + cch * 8);
        async_b128(lds_off_of(&Bs[nb][grow][gcol + cch * 8]), bgp + ko + cch * 8);
      }
    }

    const int buf = i & 1;
    #pragma unroll
    for (int s = 0; s < 2; ++s) {            // two K=32 steps per stage
      const int cb = s * 32;
      FragAB a0, a1, b0, b1, b2, b3;
      // A 16x32 fragment: lanes 0-15 hold K 0..15, lanes 16-31 hold K 16..31
      a0.h[0] = *(const bf16x8*)&As[buf][wm * 32 +  0 + lm][cb + hi * 16];
      a0.h[1] = *(const bf16x8*)&As[buf][wm * 32 +  0 + lm][cb + hi * 16 + 8];
      a1.h[0] = *(const bf16x8*)&As[buf][wm * 32 + 16 + lm][cb + hi * 16];
      a1.h[1] = *(const bf16x8*)&As[buf][wm * 32 + 16 + lm][cb + hi * 16 + 8];
      // B 32x16 fragment: lanes 0-15 hold K 0..15, lanes 16-31 hold K 16..31
      b0.h[0] = *(const bf16x8*)&Bs[buf][wn * 64 +  0 + lm][cb + hi * 16];
      b0.h[1] = *(const bf16x8*)&Bs[buf][wn * 64 +  0 + lm][cb + hi * 16 + 8];
      b1.h[0] = *(const bf16x8*)&Bs[buf][wn * 64 + 16 + lm][cb + hi * 16];
      b1.h[1] = *(const bf16x8*)&Bs[buf][wn * 64 + 16 + lm][cb + hi * 16 + 8];
      b2.h[0] = *(const bf16x8*)&Bs[buf][wn * 64 + 32 + lm][cb + hi * 16];
      b2.h[1] = *(const bf16x8*)&Bs[buf][wn * 64 + 32 + lm][cb + hi * 16 + 8];
      b3.h[0] = *(const bf16x8*)&Bs[buf][wn * 64 + 48 + lm][cb + hi * 16];
      b3.h[1] = *(const bf16x8*)&Bs[buf][wn * 64 + 48 + lm][cb + hi * 16 + 8];

      c[0][0] = __builtin_amdgcn_wmma_f32_16x16x32_bf16(false, a0.v, false, b0.v, (short)0, c[0][0], false, false);
      c[0][1] = __builtin_amdgcn_wmma_f32_16x16x32_bf16(false, a0.v, false, b1.v, (short)0, c[0][1], false, false);
      c[0][2] = __builtin_amdgcn_wmma_f32_16x16x32_bf16(false, a0.v, false, b2.v, (short)0, c[0][2], false, false);
      c[0][3] = __builtin_amdgcn_wmma_f32_16x16x32_bf16(false, a0.v, false, b3.v, (short)0, c[0][3], false, false);
      c[1][0] = __builtin_amdgcn_wmma_f32_16x16x32_bf16(false, a1.v, false, b0.v, (short)0, c[1][0], false, false);
      c[1][1] = __builtin_amdgcn_wmma_f32_16x16x32_bf16(false, a1.v, false, b1.v, (short)0, c[1][1], false, false);
      c[1][2] = __builtin_amdgcn_wmma_f32_16x16x32_bf16(false, a1.v, false, b2.v, (short)0, c[1][2], false, false);
      c[1][3] = __builtin_amdgcn_wmma_f32_16x16x32_bf16(false, a1.v, false, b3.v, (short)0, c[1][3], false, false);
    }
  }

  // C layout: accumulator element i -> row hi*8 + i, col lane&15
  #pragma unroll
  for (int a = 0; a < 2; ++a) {
    const long long crow = bm + wm * 32 + a * 16 + hi * 8;
    #pragma unroll
    for (int j = 0; j < 4; ++j) {
      const long long col = bn + wn * 64 + j * 16 + lm;
      #pragma unroll
      for (int i = 0; i < 8; ++i) {
        float v = c[a][j][i];
        const long long g = cbase + (crow + i) * (long long)N + col;
        if (bias)  v += bias[col];
        if (resid) v += resid[g];
        if (relu)  v = fmaxf(v, 0.0f);
        if (outF)  outF[g] = v;
        if (outB)  outB[g] = f32_to_bf16u(v);
      }
    }
  }
}

// ---------------------------------------------------------------------------
// Row softmax over 2048 columns (one block per row), writes bf16 attn.
// ---------------------------------------------------------------------------
__global__ __launch_bounds__(256) void softmax_rows(
    const float* __restrict__ scores, unsigned short* __restrict__ attn, int ncols)
{
  __shared__ float red[256];
  const int tid = threadIdx.x;
  const long long base = (long long)blockIdx.x * ncols;

  float vals[8];
  float m = -3.4e38f;
  #pragma unroll
  for (int i = 0; i < 8; ++i) {
    vals[i] = scores[base + tid + i * 256];
    m = fmaxf(m, vals[i]);
  }
  red[tid] = m; __syncthreads();
  for (int s = 128; s > 0; s >>= 1) {
    if (tid < s) red[tid] = fmaxf(red[tid], red[tid + s]);
    __syncthreads();
  }
  m = red[0]; __syncthreads();

  float sum = 0.f;
  #pragma unroll
  for (int i = 0; i < 8; ++i) { vals[i] = __expf(vals[i] - m); sum += vals[i]; }
  red[tid] = sum; __syncthreads();
  for (int s = 128; s > 0; s >>= 1) {
    if (tid < s) red[tid] += red[tid + s];
    __syncthreads();
  }
  const float inv = 1.0f / red[0];
  #pragma unroll
  for (int i = 0; i < 8; ++i)
    attn[base + tid + i * 256] = f32_to_bf16u(vals[i] * inv);
}

// ---------------------------------------------------------------------------
// LayerNorm over D=1024 (one block per row); writes f32 and optional bf16.
// ---------------------------------------------------------------------------
__global__ __launch_bounds__(256) void layernorm_rows(
    const float* __restrict__ in, const float* __restrict__ gamma,
    const float* __restrict__ beta, float* __restrict__ outF,
    unsigned short* __restrict__ outB, int D)
{
  __shared__ float red[256];
  const int tid = threadIdx.x;
  const long long base = (long long)blockIdx.x * D;

  float v[4];
  float s = 0.f;
  #pragma unroll
  for (int i = 0; i < 4; ++i) { v[i] = in[base + tid + i * 256]; s += v[i]; }
  red[tid] = s; __syncthreads();
  for (int k = 128; k > 0; k >>= 1) {
    if (tid < k) red[tid] += red[tid + k];
    __syncthreads();
  }
  const float mu = red[0] / (float)D; __syncthreads();

  float s2 = 0.f;
  #pragma unroll
  for (int i = 0; i < 4; ++i) { const float d = v[i] - mu; s2 += d * d; }
  red[tid] = s2; __syncthreads();
  for (int k = 128; k > 0; k >>= 1) {
    if (tid < k) red[tid] += red[tid + k];
    __syncthreads();
  }
  const float inv = rsqrtf(red[0] / (float)D + 1e-5f);

  #pragma unroll
  for (int i = 0; i < 4; ++i) {
    const int cI = tid + i * 256;
    const float o = (v[i] - mu) * inv * gamma[cI] + beta[cI];
    if (outF) outF[base + cI] = o;
    if (outB) outB[base + cI] = f32_to_bf16u(o);
  }
}

// ---------------------------------------------------------------------------
// Transpose-convert: src f32 [z][R][C] -> dst bf16 [z][C][R]. 32x32 LDS tiles.
// ---------------------------------------------------------------------------
__global__ __launch_bounds__(256) void transpose_f32_to_bf16(
    const float* __restrict__ src, unsigned short* __restrict__ dst, int R, int C)
{
  __shared__ float tile[32][33];
  const int tid = threadIdx.x;
  const int tx = tid & 31, ty = tid >> 5;     // 32 x 8 threads
  const long long zoff = (long long)blockIdx.z * (long long)R * C;
  const int c0 = blockIdx.x * 32, r0 = blockIdx.y * 32;

  #pragma unroll
  for (int j = 0; j < 4; ++j) {
    const int r = r0 + ty + j * 8;
    tile[ty + j * 8][tx] = src[zoff + (long long)r * C + c0 + tx];
  }
  __syncthreads();
  #pragma unroll
  for (int j = 0; j < 4; ++j) {
    const int cc = c0 + ty + j * 8;
    dst[zoff + (long long)cc * R + r0 + tx] = f32_to_bf16u(tile[tx][ty + j * 8]);
  }
}

// ---------------------------------------------------------------------------
// Flat f32 -> bf16 convert.
// ---------------------------------------------------------------------------
__global__ __launch_bounds__(256) void convert_f32_to_bf16(
    const float* __restrict__ in, unsigned short* __restrict__ out, long long n)
{
  long long i = (long long)blockIdx.x * blockDim.x + threadIdx.x;
  const long long stride = (long long)gridDim.x * blockDim.x;
  for (; i < n; i += stride) out[i] = f32_to_bf16u(in[i]);
}

// ---------------------------------------------------------------------------
// Host-side orchestration.
// ---------------------------------------------------------------------------
extern "C" void kernel_launch(void* const* d_in, const int* in_sizes, int n_in,
                              void* d_out, int out_size, void* d_ws, size_t ws_size,
                              hipStream_t stream)
{
  (void)in_sizes; (void)n_in; (void)out_size; (void)ws_size;

  const int  Bb = 4, Sq = 2048, Dd = 1024, Hh = 1024, Ff = 256;
  const long long Mrows = (long long)Bb * Sq;   // 8192

  const float* x   = (const float*)d_in[0];
  const float* Wq  = (const float*)d_in[1];
  const float* Wk  = (const float*)d_in[2];
  const float* Wv  = (const float*)d_in[3];
  const float* Rm  = (const float*)d_in[4];
  const float* Wo  = (const float*)d_in[5];
  const float* W1  = (const float*)d_in[6];
  const float* b1  = (const float*)d_in[7];
  const float* W2  = (const float*)d_in[8];
  const float* b2  = (const float*)d_in[9];
  const float* g1  = (const float*)d_in[10];
  const float* be1 = (const float*)d_in[11];
  const float* g2  = (const float*)d_in[12];
  const float* be2 = (const float*)d_in[13];
  float* out = (float*)d_out;

  char* ws = (char*)d_ws;
  const size_t MB = 1024ull * 1024ull;
  auto U16 = [&](size_t off) { return (unsigned short*)(ws + off); };
  auto BF  = [&](size_t off) { return (const __bf16*)(ws + off); };
  auto F32 = [&](size_t off) { return (float*)(ws + off); };

  // persistent buffers
  const size_t o_xb   = 0;          // x  bf16 [8192,1024]   16MB
  const size_t o_WqT  = 16 * MB;    // WqT bf16 [H,D]          2MB
  const size_t o_WkT  = 18 * MB;
  const size_t o_WvT  = 20 * MB;
  const size_t o_WoT  = 22 * MB;    // [D,H]
  const size_t o_W1T  = 24 * MB;    // [H,D]
  const size_t o_W2T  = 26 * MB;    // [D,H]
  const size_t o_RT   = 28 * MB;    // [F,H]                 0.5MB
  const size_t o_qb   = 32 * MB;    // q bf16 [8192,1024]    16MB
  const size_t o_kb   = 48 * MB;
  const size_t o_vT   = 64 * MB;    // vT bf16 [B][H][S]     16MB
  const size_t o_qpb  = 80 * MB;    // qp bf16 [8192,256]     4MB
  const size_t o_kpb  = 84 * MB;
  const size_t o_attn = 88 * MB;    // attn bf16 [B,S,S]     32MB
  const size_t o_outb = 120 * MB;   // out bf16 [8192,1024]  16MB
  const size_t o_x1f  = 136 * MB;   // x1 f32                32MB
  const size_t o_x1b  = 168 * MB;   // x1 bf16               16MB
  const size_t o_hb   = 184 * MB;   // relu-h bf16           16MB
  // reusable temps
  const size_t o_vf     = 200 * MB; // v f32 [8192,1024]     32MB (done after vT)
  const size_t o_y1     = 200 * MB; // x + attn_out f32      32MB (reuses vf)
  const size_t o_scores = 232 * MB; // scores f32 [B,S,S]    64MB
  const size_t o_y2     = 232 * MB; // x1 + ff f32           32MB (reuses scores)

  const dim3 t256(256);
  const long long SF = (long long)Sq * Ff;
  const long long SS = (long long)Sq * Sq;
  const long long SH = (long long)Sq * Hh;

  // 1) bf16 conversions / weight transposes
  convert_f32_to_bf16<<<8192, t256, 0, stream>>>(x, U16(o_xb), Mrows * Dd);
  transpose_f32_to_bf16<<<dim3(32, 32, 1), t256, 0, stream>>>(Wq, U16(o_WqT), Dd, Hh);
  transpose_f32_to_bf16<<<dim3(32, 32, 1), t256, 0, stream>>>(Wk, U16(o_WkT), Dd, Hh);
  transpose_f32_to_bf16<<<dim3(32, 32, 1), t256, 0, stream>>>(Wv, U16(o_WvT), Dd, Hh);
  transpose_f32_to_bf16<<<dim3(Ff / 32, Hh / 32, 1), t256, 0, stream>>>(Rm, U16(o_RT), Hh, Ff);
  transpose_f32_to_bf16<<<dim3(32, 32, 1), t256, 0, stream>>>(Wo, U16(o_WoT), Hh, Dd);
  transpose_f32_to_bf16<<<dim3(32, 32, 1), t256, 0, stream>>>(W1, U16(o_W1T), Dd, Hh);
  transpose_f32_to_bf16<<<dim3(32, 32, 1), t256, 0, stream>>>(W2, U16(o_W2T), Hh, Dd);

  // 2) q/k/v projections (q,k -> bf16 directly; v -> f32 then batch-transpose)
  gemm_bf16_wmma<<<dim3(Hh / 128, Mrows / 128, 1), t256, 0, stream>>>(
      BF(o_xb), BF(o_WqT), nullptr, nullptr, nullptr, U16(o_qb),
      (int)Mrows, Hh, Dd, 0, 0, 0, 0);
  gemm_bf16_wmma<<<dim3(Hh / 128, Mrows / 128, 1), t256, 0, stream>>>(
      BF(o_xb), BF(o_WkT), nullptr, nullptr, nullptr, U16(o_kb),
      (int)Mrows, Hh, Dd, 0, 0, 0, 0);
  gemm_bf16_wmma<<<dim3(Hh / 128, Mrows / 128, 1), t256, 0, stream>>>(
      BF(o_xb), BF(o_WvT), nullptr, nullptr, F32(o_vf), nullptr,
      (int)Mrows, Hh, Dd, 0, 0, 0, 0);
  transpose_f32_to_bf16<<<dim3(Hh / 32, Sq / 32, Bb), t256, 0, stream>>>(
      F32(o_vf), U16(o_vT), Sq, Hh);

  // 3) R projections
  gemm_bf16_wmma<<<dim3(Ff / 128, Mrows / 128, 1), t256, 0, stream>>>(
      BF(o_qb), BF(o_RT), nullptr, nullptr, nullptr, U16(o_qpb),
      (int)Mrows, Ff, Hh, 0, 0, 0, 0);
  gemm_bf16_wmma<<<dim3(Ff / 128, Mrows / 128, 1), t256, 0, stream>>>(
      BF(o_kb), BF(o_RT), nullptr, nullptr, nullptr, U16(o_kpb),
      (int)Mrows, Ff, Hh, 0, 0, 0, 0);

  // 4) scores[b] = qp[b] . kp[b]^T   (batched, kp already in [N,K] layout)
  gemm_bf16_wmma<<<dim3(Sq / 128, Sq / 128, Bb), t256, 0, stream>>>(
      BF(o_qpb), BF(o_kpb), nullptr, nullptr, F32(o_scores), nullptr,
      Sq, Sq, Ff, SF, SF, SS, 0);

  // 5) softmax -> bf16 attention
  softmax_rows<<<Bb * Sq, t256, 0, stream>>>(F32(o_scores), U16(o_attn), Sq);

  // 6) out[b] = attn[b] . v[b]   (vT is [H,S] per batch)
  gemm_bf16_wmma<<<dim3(Hh / 128, Sq / 128, Bb), t256, 0, stream>>>(
      BF(o_attn), BF(o_vT), nullptr, nullptr, nullptr, U16(o_outb),
      Sq, Hh, Sq, SS, SH, SH, 0);

  // 7) y1 = out . Wo + x (residual fused)
  gemm_bf16_wmma<<<dim3(Dd / 128, Mrows / 128, 1), t256, 0, stream>>>(
      BF(o_outb), BF(o_WoT), nullptr, x, F32(o_y1), nullptr,
      (int)Mrows, Dd, Hh, 0, 0, 0, 0);

  // 8) x1 = LN(y1)
  layernorm_rows<<<(int)Mrows, t256, 0, stream>>>(
      F32(o_y1), g1, be1, F32(o_x1f), U16(o_x1b), Dd);

  // 9) h = relu(x1 . W1 + b1) -> bf16
  gemm_bf16_wmma<<<dim3(Hh / 128, Mrows / 128, 1), t256, 0, stream>>>(
      BF(o_x1b), BF(o_W1T), b1, nullptr, nullptr, U16(o_hb),
      (int)Mrows, Hh, Dd, 0, 0, 0, 1);

  // 10) y2 = h . W2 + b2 + x1 (bias + residual fused)
  gemm_bf16_wmma<<<dim3(Dd / 128, Mrows / 128, 1), t256, 0, stream>>>(
      BF(o_hb), BF(o_W2T), b2, F32(o_x1f), F32(o_y2), nullptr,
      (int)Mrows, Dd, Hh, 0, 0, 0, 0);

  // 11) final LN -> d_out
  layernorm_rows<<<(int)Mrows, t256, 0, stream>>>(
      F32(o_y2), g2, be2, out, nullptr, Dd);
}
